// CausalSelfAttention_39943195853480
// MI455X (gfx1250) — matrix operand order repaired and checked by reading, not compiled, not run on hardware
//
#include <hip/hip_runtime.h>

// ---------------------------------------------------------------------------
// CDNA5 / gfx1250 causal self-attention:  y = proj( softmax(rope(q)·rope(k)ᵀ)·v )
// All matmuls on v_wmma_f32_16x16x32_bf16 (wave32).
// ---------------------------------------------------------------------------

typedef __attribute__((ext_vector_type(16))) __bf16 v16bf;
typedef __attribute__((ext_vector_type(8)))  float  v8f;

#define WMMA_BF16(a, b, c) \
  __builtin_amdgcn_wmma_f32_16x16x32_bf16(false, (a), false, (b), (short)0, (c), false, false)

union AFrag { v16bf v; uint4 q[2]; };
union CFrag { v8f   v; float e[8]; };

__device__ __forceinline__ float bf2f(const unsigned short* p) {
  return (float)(*(const __bf16*)p);
}
__device__ __forceinline__ void f2bf(unsigned short* p, float f) {
  *(__bf16*)p = (__bf16)f;
}

// ---------------------------------------------------------------------------
// fp32 -> bf16 conversion (4 elements/thread, n % 4 == 0)
// ---------------------------------------------------------------------------
__global__ __launch_bounds__(256) void cvt_f32_bf16_kernel(
    const float* __restrict__ in, unsigned short* __restrict__ out, int n) {
  int i = (blockIdx.x * 256 + threadIdx.x) * 4;
  if (i >= n) return;
  float4 f = *(const float4*)(in + i);
  __bf16* o = (__bf16*)(out + i);
  o[0] = (__bf16)f.x; o[1] = (__bf16)f.y; o[2] = (__bf16)f.z; o[3] = (__bf16)f.w;
}

// ---------------------------------------------------------------------------
// Tiled GEMM: C(MxN) = A(MxK,bf16,row-major) * B(KxN,bf16,row-major)
// 128x128 tile per 256-thread block (8 waves), K-step 32.
// ---------------------------------------------------------------------------
template <bool BF16_OUT>
__global__ __launch_bounds__(256) void gemm_bf16_wmma_kernel(
    const unsigned short* __restrict__ Au, const unsigned short* __restrict__ Bu,
    void* __restrict__ Cout, int M, int N, int K) {
  const __bf16* A = (const __bf16*)Au;
  const __bf16* B = (const __bf16*)Bu;

  __shared__ __bf16 As[128][32];      // 8 KB, row-major MxK tile
  __shared__ __bf16 Bs[128][40];      // 10 KB, transposed (N-major) + pad, rows 80B (16B mult.)

  const int tid  = threadIdx.x;
  // wave index is wave-uniform: force into SGPR so LDS base math is scalar.
  const int wave = __builtin_amdgcn_readfirstlane(tid >> 5);
  const int lane = tid & 31;
  const int half = lane >> 4;
  const int l16  = lane & 15;
  const int m0   = blockIdx.y * 128;
  const int n0   = blockIdx.x * 128;

  CFrag acc[8];
#pragma unroll
  for (int j = 0; j < 8; ++j) acc[j].v = (v8f){};

  for (int k0 = 0; k0 < K; k0 += 32) {
    __syncthreads();
    // --- A tile: 128x32, 16 bf16 per thread (2 x b128) ---
    {
      int row = tid >> 1;
      int cb  = (tid & 1) * 16;
      const uint4* src = (const uint4*)(A + (size_t)(m0 + row) * K + k0 + cb);
      uint4 d0 = src[0];
      uint4 d1 = src[1];
      *(uint4*)(&As[row][cb])     = d0;
      *(uint4*)(&As[row][cb + 8]) = d1;
    }
    // --- B tile: 32x128, stored transposed into Bs[n][k] ---
#pragma unroll
    for (int j = 0; j < 2; ++j) {
      int c  = tid + j * 256;          // 0..511 chunks of 8
      int kk = c >> 4;                 // 0..31
      int nb = (c & 15) * 8;           // 0..120
      uint4 d = *(const uint4*)(B + (size_t)(k0 + kk) * N + n0 + nb);
      const __bf16* e = (const __bf16*)&d;
#pragma unroll
      for (int i = 0; i < 8; ++i) Bs[nb + i][kk] = e[i];
    }
    if (k0 + 32 < K) {  // prefetch next A tile row for this thread
      __builtin_prefetch(A + (size_t)(m0 + (tid >> 1)) * K + k0 + 32, 0, 0);
    }
    __syncthreads();

    // --- compute: each wave does a 16x128 strip = 8 WMMAs ---
    AFrag a;
    const int arow = wave * 16 + l16;
    a.q[0] = *(const uint4*)(&As[arow][half * 8]);        // K {0..7}/{8..15}
    a.q[1] = *(const uint4*)(&As[arow][16 + half * 8]);   // K {16..23}/{24..31}
#pragma unroll
    for (int j = 0; j < 8; ++j) {
      AFrag b;
      const int bcol = j * 16 + l16;
      b.q[0] = *(const uint4*)(&Bs[bcol][half * 8]);      // K {0..7}/{8..15}
      b.q[1] = *(const uint4*)(&Bs[bcol][16 + half * 8]); // K {16..23}/{24..31}
      acc[j].v = WMMA_BF16(a.v, b.v, acc[j].v);
    }
  }

  // --- epilogue: D layout -> memory ---
#pragma unroll
  for (int j = 0; j < 8; ++j) {
#pragma unroll
    for (int r = 0; r < 8; ++r) {
      const int row = m0 + wave * 16 + half * 8 + r;
      const int col = n0 + j * 16 + l16;
      if (BF16_OUT) {
        ((__bf16*)Cout)[(size_t)row * N + col] = (__bf16)acc[j].e[r];
      } else {
        ((float*)Cout)[(size_t)row * N + col] = acc[j].e[r];
      }
    }
  }
}

// ---------------------------------------------------------------------------
// RoPE + head split.  qkv: [B*T, 3*768] bf16 -> q,k,v: [B*H, T, 64] bf16.
// One thread per (bt, head, even/odd pair).
// ---------------------------------------------------------------------------
__global__ __launch_bounds__(256) void rope_split_kernel(
    const unsigned short* __restrict__ qkv, unsigned short* __restrict__ q,
    unsigned short* __restrict__ k, unsigned short* __restrict__ v) {
  const int idx = blockIdx.x * 256 + threadIdx.x;  // B*T * 12 * 32
  const int j  = idx & 31;
  const int h  = (idx >> 5) % 12;
  const int bt = idx / (32 * 12);
  const int t  = bt & 2047;
  const int b  = bt >> 11;

  const unsigned short* row = qkv + (size_t)bt * 2304;
  const int de = 2 * j, dodd = 2 * j + 1;

  const float qe = bf2f(row + h * 64 + de);
  const float qo = bf2f(row + h * 64 + dodd);
  const float ke = bf2f(row + 768 + h * 64 + de);
  const float ko = bf2f(row + 768 + h * 64 + dodd);
  const float ve = bf2f(row + 1536 + h * 64 + de);
  const float vo = bf2f(row + 1536 + h * 64 + dodd);

  // theta_d = t * 10000^(-(d % 32)/32);  ln(10000)/32 = 0.28782313662...
  const float c0 = 0.2878231366242712f;
  const float fe = __expf(-(float)(de & 31) * c0);
  const float fo = __expf(-(float)(dodd & 31) * c0);
  const float ae = (float)t * fe, ao = (float)t * fo;
  const float ce = __cosf(ae), se = __sinf(ae);
  const float co = __cosf(ao), so = __sinf(ao);

  // out[2i] = x[2i]*cos - x[2i+1]*sin ; out[2i+1] = x[2i+1]*cos + x[2i]*sin
  const size_t dst = ((size_t)(b * 12 + h) * 2048 + t) * 64;
  f2bf(q + dst + de,   qe * ce - qo * se);
  f2bf(q + dst + dodd, qo * co + qe * so);
  f2bf(k + dst + de,   ke * ce - ko * se);
  f2bf(k + dst + dodd, ko * co + ke * so);
  f2bf(v + dst + de,   ve);
  f2bf(v + dst + dodd, vo);
}

// ---------------------------------------------------------------------------
// Flash attention (causal).  q,k,v: [B*H, 2048, 64] bf16.
// Block = 256 threads = 8 waves; each wave owns a 16-row Q tile (128 rows/block).
// K streamed in 32-key chunks through LDS; V stored transposed in LDS.
// Output: [B, T, C] bf16 (heads re-merged) for the projection GEMM.
// ---------------------------------------------------------------------------
__global__ __launch_bounds__(256) void flash_attn_kernel(
    const unsigned short* __restrict__ Qu, const unsigned short* __restrict__ Ku,
    const unsigned short* __restrict__ Vu, unsigned short* __restrict__ Ou) {
  const __bf16* Q = (const __bf16*)Qu;
  const __bf16* K = (const __bf16*)Ku;
  const __bf16* V = (const __bf16*)Vu;

  __shared__ __bf16 Ks[32][64];    // 4 KB, key-major
  __shared__ __bf16 Vt[64][32];    // 4 KB, transposed (d-major)
  __shared__ __bf16 Ps[8][16][32]; // 8 KB, per-wave P staging (D-layout -> A-layout)

  const int tid  = threadIdx.x;
  // wave index is wave-uniform: keep it (and everything derived) in SGPRs.
  const int wave = __builtin_amdgcn_readfirstlane(tid >> 5);
  const int lane = tid & 31;
  const int half = lane >> 4;
  const int l16  = lane & 15;
  const int bh   = blockIdx.y;     // b*12 + h
  const int q0   = blockIdx.x * 128;
  const int qw   = q0 + wave * 16; // this wave's first Q row (scalar)
  const size_t base = (size_t)bh * 2048 * 64;

  // --- load Q fragments once (A layout, K=d split 0..31 / 32..63) ---
  AFrag aq0, aq1;
  {
    const __bf16* qrow = Q + base + (size_t)(qw + l16) * 64;
    const int kb = half * 8;
    aq0.q[0] = *(const uint4*)(qrow + kb);
    aq0.q[1] = *(const uint4*)(qrow + 16 + kb);
    aq1.q[0] = *(const uint4*)(qrow + 32 + kb);
    aq1.q[1] = *(const uint4*)(qrow + 48 + kb);
  }

  CFrag o[4];
#pragma unroll
  for (int t = 0; t < 4; ++t) o[t].v = (v8f){};
  float m_i[8], l_i[8];
#pragma unroll
  for (int r = 0; r < 8; ++r) { m_i[r] = -1e30f; l_i[r] = 0.f; }

  const int nchunk = (q0 + 128) >> 5;
  for (int c = 0; c < nchunk; ++c) {
    const int k0 = c * 32;
    __syncthreads();  // protect Ks/Vt (and Ps) against overwrite
    {
      const int r  = tid >> 3;        // key 0..31
      const int cb = (tid & 7) * 8;   // d offset
      uint4 kd = *(const uint4*)(K + base + (size_t)(k0 + r) * 64 + cb);
      *(uint4*)(&Ks[r][cb]) = kd;
      uint4 vd = *(const uint4*)(V + base + (size_t)(k0 + r) * 64 + cb);
      const __bf16* e = (const __bf16*)&vd;
#pragma unroll
      for (int i = 0; i < 8; ++i) Vt[cb + i][r] = e[i];
    }
    __syncthreads();

    // Wave-uniform chunk guard, forced scalar so the compiler emits
    // s_cmp/s_cbranch (EXEC stays all-ones around every v_wmma, per ISA).
    const bool active = __builtin_amdgcn_readfirstlane((int)(k0 <= qw + 15)) != 0;
    if (active) {
      // --- S = Q * K^T  (16 x 32 scores, two 16x16 N-tiles, K=64 = 2x32) ---
      CFrag s[2];
#pragma unroll
      for (int nt = 0; nt < 2; ++nt) {
        AFrag b0, b1;
        const __bf16* krow = &Ks[nt * 16 + l16][0];
        b0.q[0] = *(const uint4*)(krow + half * 8);
        b0.q[1] = *(const uint4*)(krow + 16 + half * 8);
        b1.q[0] = *(const uint4*)(krow + 32 + half * 8);
        b1.q[1] = *(const uint4*)(krow + 48 + half * 8);
        s[nt].v = (v8f){};
        s[nt].v = WMMA_BF16(aq0.v, b0.v, s[nt].v);
        s[nt].v = WMMA_BF16(aq1.v, b1.v, s[nt].v);
      }

      // --- online softmax (rows live in 16-lane halves; width-16 reductions) ---
      float alpha[8];
#pragma unroll
      for (int r = 0; r < 8; ++r) {
        const int qg = qw + half * 8 + r;
        float s0 = s[0].e[r] * 0.125f;   // 1/sqrt(64)
        float s1 = s[1].e[r] * 0.125f;
        const bool msk0 = (k0 + l16) > qg;
        const bool msk1 = (k0 + 16 + l16) > qg;
        if (msk0) s0 = -1e30f;
        if (msk1) s1 = -1e30f;
        float vmax = fmaxf(s0, s1);
#pragma unroll
        for (int off = 1; off < 16; off <<= 1)
          vmax = fmaxf(vmax, __shfl_xor(vmax, off, 16));
        const float mnew = fmaxf(m_i[r], vmax);
        const float a    = __expf(m_i[r] - mnew);
        const float p0   = msk0 ? 0.f : __expf(s0 - mnew);
        const float p1   = msk1 ? 0.f : __expf(s1 - mnew);
        float rs = p0 + p1;
#pragma unroll
        for (int off = 1; off < 16; off <<= 1)
          rs += __shfl_xor(rs, off, 16);
        m_i[r]   = mnew;
        l_i[r]   = l_i[r] * a + rs;
        alpha[r] = a;
        const int prow = half * 8 + r;
        Ps[wave][prow][l16]      = (__bf16)p0;
        Ps[wave][prow][16 + l16] = (__bf16)p1;
      }
#pragma unroll
      for (int t = 0; t < 4; ++t)
#pragma unroll
        for (int r = 0; r < 8; ++r) o[t].e[r] *= alpha[r];
    }
    __syncthreads();  // Ps visible (wave-local, but keeps barriers uniform)

    if (active) {
      // --- O += P * V  (A = P 16x32, B = V 32x64 -> four 16x16 tiles) ---
      AFrag ap;
      const __bf16* prow = &Ps[wave][l16][0];
      ap.q[0] = *(const uint4*)(prow + half * 8);
      ap.q[1] = *(const uint4*)(prow + 16 + half * 8);
#pragma unroll
      for (int t = 0; t < 4; ++t) {
        AFrag bv;
        const __bf16* vrow = &Vt[t * 16 + l16][0];
        bv.q[0] = *(const uint4*)(vrow + half * 8);
        bv.q[1] = *(const uint4*)(vrow + 16 + half * 8);
        o[t].v = WMMA_BF16(ap.v, bv.v, o[t].v);
      }
    }
  }

  // --- normalize and write merged-head output [B, T, C] bf16 ---
  const int b = bh / 12, h = bh % 12;
#pragma unroll
  for (int t = 0; t < 4; ++t)
#pragma unroll
    for (int r = 0; r < 8; ++r) {
      const int qrow = qw + half * 8 + r;
      const float val = o[t].e[r] / l_i[r];
      ((__bf16*)Ou)[((size_t)(b * 2048 + qrow)) * 768 + h * 64 + t * 16 + l16] =
          (__bf16)val;
    }
}

// ---------------------------------------------------------------------------
// Host launch
// ---------------------------------------------------------------------------
extern "C" void kernel_launch(void* const* d_in, const int* in_sizes, int n_in,
                              void* d_out, int out_size, void* d_ws, size_t ws_size,
                              hipStream_t stream) {
  const float* x      = (const float*)d_in[0];  // [4,2048,768]
  const float* w_attn = (const float*)d_in[1];  // [768,2304]
  const float* w_proj = (const float*)d_in[2];  // [768,768]
  float* out = (float*)d_out;                   // [4,2048,768]

  const size_t BT = 8192, C = 768;
  char* p = (char*)d_ws;
  unsigned short* xb   = (unsigned short*)p; p += BT * C * 2;          // 12.6 MB
  unsigned short* wab  = (unsigned short*)p; p += C * 3 * C * 2;       //  3.5 MB
  unsigned short* wpb  = (unsigned short*)p; p += C * C * 2;           //  1.2 MB
  unsigned short* qkvb = (unsigned short*)p; p += BT * 3 * C * 2;      // 37.7 MB
  unsigned short* qb   = (unsigned short*)p; p += BT * C * 2;          // 12.6 MB
  unsigned short* kb   = (unsigned short*)p; p += BT * C * 2;          // 12.6 MB
  unsigned short* vb   = (unsigned short*)p; p += BT * C * 2;          // 12.6 MB
  unsigned short* attb = (unsigned short*)p; p += BT * C * 2;          // 12.6 MB

  // 1) fp32 -> bf16
  cvt_f32_bf16_kernel<<<(BT * C) / 1024, 256, 0, stream>>>(x, xb, (int)(BT * C));
  cvt_f32_bf16_kernel<<<(C * 3 * C) / 1024, 256, 0, stream>>>(w_attn, wab, (int)(C * 3 * C));
  cvt_f32_bf16_kernel<<<(C * C) / 1024, 256, 0, stream>>>(w_proj, wpb, (int)(C * C));

  // 2) qkv = x @ w_attn   (8192 x 2304 x 768)
  gemm_bf16_wmma_kernel<true><<<dim3(2304 / 128, 8192 / 128), 256, 0, stream>>>(
      xb, wab, qkvb, 8192, 2304, 768);

  // 3) RoPE + head split
  rope_split_kernel<<<(BT * 12 * 32) / 256, 256, 0, stream>>>(qkvb, qb, kb, vb);

  // 4) causal flash attention
  flash_attn_kernel<<<dim3(2048 / 128, 48), 256, 0, stream>>>(qb, kb, vb, attb);

  // 5) y = att @ w_proj   (8192 x 768 x 768), fp32 out
  gemm_bf16_wmma_kernel<false><<<dim3(768 / 128, 8192 / 128), 256, 0, stream>>>(
      attb, wpb, out, 8192, 768, 768);
}
